// VectorModule_87505663689249
// MI455X (gfx1250) — compile-verified, matrix operand order checked by reading, not run on hardware
//
#include <hip/hip_runtime.h>
#include <hip/hip_bf16.h>
#include <math.h>

// ---------------------------------------------------------------------------
// Sizes from the reference: N=768, B=16 rbf basis, H=32 hidden, C=32 channels.
// Heavy cost = per-pair filter MLPs (7 filters x N^2 x (16->32->32)) ~ 12.7 GF,
// mapped onto V_WMMA_F32_16X16X32_F16 (16 pairs per tile per wave).
// Fast v_rcp_f32 / v_sqrt_f32 used instead of IEEE div/sqrt sequences.
// ---------------------------------------------------------------------------
#define N_PTS 768
#define NB 16
#define CCH 32
#define HD 32
#define EPSF 1e-7f
#define RCUT 5.0f

typedef __attribute__((ext_vector_type(16))) _Float16 v16h;
typedef __attribute__((ext_vector_type(8)))  _Float16 v8h;
typedef __attribute__((ext_vector_type(8)))  float    v8f;

__device__ __forceinline__ float frcp(float x) {
#if __has_builtin(__builtin_amdgcn_rcpf)
  return __builtin_amdgcn_rcpf(x);        // single v_rcp_f32
#else
  return 1.0f / x;
#endif
}

__device__ __forceinline__ float fsqrt(float x) {
#if __has_builtin(__builtin_amdgcn_sqrtf)
  return __builtin_amdgcn_sqrtf(x);       // single v_sqrt_f32
#else
  return sqrtf(x);
#endif
}

__device__ __forceinline__ float swishf(float x) {
  return x * frcp(1.0f + __expf(-x));
}

// B-operand loader from LDS (weights stored transposed: [n][K]), K=16 padded
// to 32 or K=32 native, then one WMMA. Lane n-column = (n arg), K split per
// the gfx1250 16-bit operand layout (lanes 0-15: K 0..7 / 16..23, lanes
// 16-31: K 8..15 / 24..31 across the two v8h halves).
template<int K>
__device__ __forceinline__ v8f wmma_bt(v16h a, const _Float16* bt, int n, int hi, v8f c) {
  v16h b;
  if (K == 16) {
    const v8h lo = *reinterpret_cast<const v8h*>(bt + n*16 + hi*8);
#pragma unroll
    for (int e = 0; e < 8; ++e) { b[e] = lo[e]; b[e+8] = (_Float16)0.0f; }
  } else {
    const v8h lo = *reinterpret_cast<const v8h*>(bt + n*32 + hi*8);
    const v8h hv = *reinterpret_cast<const v8h*>(bt + n*32 + 16 + hi*8);
#pragma unroll
    for (int e = 0; e < 8; ++e) { b[e] = lo[e]; b[e+8] = hv[e]; }
  }
  return __builtin_amdgcn_wmma_f32_16x16x32_f16(false, a, false, b, (short)0, c,
                                                false, false);
}

// ---------------------------------------------------------------------------
// Pairwise geometry: unit vectors (f32) and masked rbf basis (f16).
// ---------------------------------------------------------------------------
__global__ __launch_bounds__(256)
void geom_kernel(const float* __restrict__ y, _Float16* __restrict__ rbf,
                 float* __restrict__ unit) {
  const size_t pid = (size_t)blockIdx.x*256 + threadIdx.x;
  if (pid >= (size_t)N_PTS*N_PTS) return;
  const int i = (int)(pid / N_PTS), j = (int)(pid % N_PTS);
  const float dx = y[i*3+0]-y[j*3+0];
  const float dy = y[i*3+1]-y[j*3+1];
  const float dz = y[i*3+2]-y[j*3+2];
  const float mask = (i == j) ? 0.0f : 1.0f;
  const float nrm = fsqrt(dx*dx+dy*dy+dz*dz + EPSF) * mask;
  const float inv = frcp(nrm + EPSF);
  unit[pid*3+0] = dx*inv;
  unit[pid*3+1] = dy*inv;
  unit[pid*3+2] = dz*inv;
  const float s = nrm * (3.14159265358979323846f / RCUT);
  _Float16* rp = rbf + pid*NB;
#pragma unroll
  for (int k = 0; k < NB; ++k)
    rp[k] = (_Float16)(__sinf((float)(k+1)*s) * inv * mask);
}

// ---------------------------------------------------------------------------
// time-embedding MLP (1->32->16) + f0 = [feat0 | temb]
// ---------------------------------------------------------------------------
__global__ __launch_bounds__(256)
void f0_kernel(const float* __restrict__ t, const float* __restrict__ feat0,
               const float* __restrict__ tW1, const float* __restrict__ tW2,
               const float* __restrict__ tb1, const float* __restrict__ tb2,
               float* __restrict__ f0) {
  __shared__ float sh[HD];
  __shared__ float stemb[16];
  const float tv = t[0];
  if (threadIdx.x < HD) sh[threadIdx.x] = swishf(tv*tW1[threadIdx.x] + tb1[threadIdx.x]);
  __syncthreads();
  if (threadIdx.x < 16) {
    float a = tb2[threadIdx.x];
    for (int c = 0; c < HD; ++c) a += sh[c]*tW2[c*16 + threadIdx.x];
    stemb[threadIdx.x] = a;
  }
  __syncthreads();
  for (int idx = blockIdx.x*256 + threadIdx.x; idx < N_PTS*CCH; idx += gridDim.x*256) {
    const int row = idx / CCH, col = idx % CCH;
    f0[idx] = (col < 16) ? feat0[row*16 + col] : stemb[col-16];
  }
}

// ---------------------------------------------------------------------------
// Fused conv layer: per block = one target node i; 8 waves each own j-tiles
// of 16 pairs. Filter MLP via WMMA, combine+row-reduce in accumulator layout.
// Filter order (= pytree flatten order): 0=p00 1=p01 2=p10 3=p11c 4=p11i.
// ---------------------------------------------------------------------------
struct ConvP {
  const float* W1[5];
  const float* W2[5];
  const float* b1[5];
  const float* b2[5];
};

template<int L>
__global__ __launch_bounds__(256)
void conv_kernel(const _Float16* __restrict__ rbf,
                 const float* __restrict__ unit,
                 const float* __restrict__ f0g,
                 const float* __restrict__ f1g,
                 ConvP P,
                 float* __restrict__ c0,
                 float* __restrict__ c1) {
  constexpr int NF = (L == 0) ? 2 : 5;
  constexpr int R0 = (L == 0) ? 32 : 64;     // c0 channels
  constexpr int R1 = (L == 0) ? 96 : 288;    // c1 channels*3

  __shared__ _Float16 w1t[NF][HD*16];        // transposed [n][k], K=16
  __shared__ _Float16 w2t[NF][CCH*32];       // transposed [n][k], K=32
  __shared__ float b1s[NF][HD];
  __shared__ float b2s[NF][CCH];
  __shared__ _Float16 hscr[8][16*32];        // per-wave transpose scratch
  __shared__ float red0[R0];
  __shared__ float red1[R1];

  const int tid = threadIdx.x;
  const int i = blockIdx.x;

  // stage weights f32->f16 transposed into LDS
  for (int idx = tid; idx < NF*512; idx += 256) {
    const int f = idx >> 9, r = idx & 511;
    const int n = r >> 4, k = r & 15;
    w1t[f][r] = (_Float16)P.W1[f][k*HD + n];
  }
  for (int idx = tid; idx < NF*1024; idx += 256) {
    const int f = idx >> 10, r = idx & 1023;
    const int n = r >> 5, k = r & 31;
    w2t[f][r] = (_Float16)P.W2[f][k*CCH + n];
  }
  for (int idx = tid; idx < NF*32; idx += 256) {
    b1s[idx>>5][idx&31] = P.b1[idx>>5][idx&31];
    b2s[idx>>5][idx&31] = P.b2[idx>>5][idx&31];
  }
  for (int idx = tid; idx < R0; idx += 256) red0[idx] = 0.0f;
  for (int idx = tid; idx < R1; idx += 256) red1[idx] = 0.0f;
  __syncthreads();

  const int wid = tid >> 5;
  const int lane = tid & 31;
  const int m16 = lane & 15;
  const int hi = lane >> 4;
  _Float16* hs = hscr[wid];

  float r00[2]     = {0.0f, 0.0f};
  float r01[2][3]  = {};
  float r10[2]     = {0.0f, 0.0f};
  float r11i[2][3] = {};
  float r11c[2][3] = {};

  const _Float16* rbf_row = rbf + (size_t)i*N_PTS*NB;

  for (int tile = wid; tile < N_PTS/16; tile += 8) {
    const int jbase = tile * 16;
    // A operand: 16 pairs x K16 of rbf (zero-pad K to 32)
    v16h arbf;
    {
      const v8h lo = *reinterpret_cast<const v8h*>(
          rbf_row + ((size_t)jbase + m16)*NB + hi*8);
#pragma unroll
      for (int e = 0; e < 8; ++e) { arbf[e] = lo[e]; arbf[e+8] = (_Float16)0.0f; }
    }
    // prefetch next tile for this wave (global_prefetch_b8)
    if (tile + 8 < N_PTS/16)
      __builtin_prefetch(rbf_row + ((size_t)(jbase + 128) + m16)*NB, 0, 1);

#pragma unroll
    for (int f = 0; f < NF; ++f) {
      // GEMM1: hidden = rbf @ W1 (16x32), then fused bias+swish
      v8f h0 = {}, h1 = {};
      h0 = wmma_bt<16>(arbf, w1t[f], m16,      hi, h0);
      h1 = wmma_bt<16>(arbf, w1t[f], 16 + m16, hi, h1);
      const float b1a = b1s[f][m16];
      const float b1b = b1s[f][16 + m16];
#pragma unroll
      for (int e = 0; e < 8; ++e) {
        h0[e] = swishf(h0[e] + b1a);
        h1[e] = swishf(h1[e] + b1b);
      }
      // C-layout -> A-layout transpose through per-wave LDS scratch
      const int rowb = hi*8;
#pragma unroll
      for (int e = 0; e < 8; ++e) {
        hs[(rowb+e)*32 + m16]      = (_Float16)h0[e];
        hs[(rowb+e)*32 + 16 + m16] = (_Float16)h1[e];
      }
      asm volatile("s_wait_dscnt 0" ::: "memory");
      v16h a2;
      {
        const v8h lo = *reinterpret_cast<const v8h*>(hs + m16*32 + hi*8);
        const v8h hv = *reinterpret_cast<const v8h*>(hs + m16*32 + 16 + hi*8);
#pragma unroll
        for (int e = 0; e < 8; ++e) { a2[e] = lo[e]; a2[e+8] = hv[e]; }
      }
      asm volatile("" ::: "memory");
      // GEMM2: F = hidden @ W2 (16x32)
      v8f F0 = {}, F1 = {};
      F0 = wmma_bt<32>(a2, w2t[f], m16,      hi, F0);
      F1 = wmma_bt<32>(a2, w2t[f], 16 + m16, hi, F1);

      // combine with f0/f1/unit and reduce over the 16 rows (8 per lane-half)
#pragma unroll
      for (int hsel = 0; hsel < 2; ++hsel) {
        const v8f F = hsel ? F1 : F0;
        const int col = hsel*16 + m16;
        const float b2v = b2s[f][col];
        const int jr0 = jbase + hi*8;
#pragma unroll
        for (int v = 0; v < 8; ++v) {
          const int j = jr0 + v;
          const float Fv = (j == i) ? 0.0f : (F[v] + b2v);  // bias + diag mask
          if (f == 0) {                       // p00: sum_j F*f0
            r00[hsel] += Fv * f0g[j*CCH + col];
          } else if (f == 1) {                // p01: sum_j F*f0*u_m
            const float tt = Fv * f0g[j*CCH + col];
            const float* up = unit + ((size_t)i*N_PTS + j)*3;
            r01[hsel][0] += tt * up[0];
            r01[hsel][1] += tt * up[1];
            r01[hsel][2] += tt * up[2];
          } else if (f == 2) {                // p10: sum_j F*(f1 . u)
            const float* up = unit + ((size_t)i*N_PTS + j)*3;
            const float* fp = f1g + ((size_t)j*CCH + col)*3;
            const float d = fp[0]*up[0] + fp[1]*up[1] + fp[2]*up[2];
            r10[hsel] += Fv * d;
          } else if (f == 3) {                // p11c: sum_j F*(u x f1)
            const float* up = unit + ((size_t)i*N_PTS + j)*3;
            const float* fp = f1g + ((size_t)j*CCH + col)*3;
            r11c[hsel][0] += Fv * (up[1]*fp[2] - up[2]*fp[1]);
            r11c[hsel][1] += Fv * (up[2]*fp[0] - up[0]*fp[2]);
            r11c[hsel][2] += Fv * (up[0]*fp[1] - up[1]*fp[0]);
          } else {                            // f == 4, p11i: sum_j F*f1_m
            const float* fp = f1g + ((size_t)j*CCH + col)*3;
            r11i[hsel][0] += Fv * fp[0];
            r11i[hsel][1] += Fv * fp[1];
            r11i[hsel][2] += Fv * fp[2];
          }
        }
      }
    }
  }

  // fold the two row-halves (lane ^ 16) and accumulate across waves
#pragma unroll
  for (int hsel = 0; hsel < 2; ++hsel) {
    const int col = hsel*16 + m16;
    float tt;
    tt = r00[hsel] + __shfl_xor(r00[hsel], 16, 32);
    if (lane < 16) atomicAdd(&red0[col], tt);
#pragma unroll
    for (int m = 0; m < 3; ++m) {
      tt = r01[hsel][m] + __shfl_xor(r01[hsel][m], 16, 32);
      if (lane < 16) atomicAdd(&red1[col*3 + m], tt);
    }
    if (L == 1) {
      tt = r10[hsel] + __shfl_xor(r10[hsel], 16, 32);
      if (lane < 16) atomicAdd(&red0[32 + col], tt);
#pragma unroll
      for (int m = 0; m < 3; ++m) {
        tt = r11i[hsel][m] + __shfl_xor(r11i[hsel][m], 16, 32);
        if (lane < 16) atomicAdd(&red1[(32 + col)*3 + m], tt);
        tt = r11c[hsel][m] + __shfl_xor(r11c[hsel][m], 16, 32);
        if (lane < 16) atomicAdd(&red1[(64 + col)*3 + m], tt);
      }
    }
  }
  __syncthreads();
  for (int idx = tid; idx < R0; idx += 256) c0[(size_t)i*R0 + idx] = red0[idx];
  for (int idx = tid; idx < R1; idx += 256) c1[(size_t)i*R1 + idx] = red1[idx];
}

// ---------------------------------------------------------------------------
// Per-node scalar MLP (c0 -> swish(H) -> C). One wave per node.
// ---------------------------------------------------------------------------
template<int CIN>
__global__ __launch_bounds__(256)
void mlpL0_kernel(const float* __restrict__ x, const float* __restrict__ W1,
                  const float* __restrict__ b1, const float* __restrict__ W2,
                  const float* __restrict__ b2, float* __restrict__ out) {
  __shared__ float hsh[8][HD];
  const int wid = threadIdx.x >> 5, lane = threadIdx.x & 31;
  const int node = blockIdx.x*8 + wid;
  const float* xr = x + (size_t)node*CIN;
  float a = b1[lane];
  for (int c = 0; c < CIN; ++c) a += xr[c]*W1[c*HD + lane];
  hsh[wid][lane] = swishf(a);
  __syncthreads();
  float o = b2[lane];
  for (int h = 0; h < HD; ++h) o += hsh[wid][h]*W2[h*CCH + lane];
  out[(size_t)node*CCH + lane] = o;
}

// ---------------------------------------------------------------------------
// Per-node equivariant MLP with gated norm. One wave per node.
// ---------------------------------------------------------------------------
template<int CIN, int DOUT>
__global__ __launch_bounds__(256)
void mlpL1_kernel(const float* __restrict__ x, const float* __restrict__ W1,
                  const float* __restrict__ b1, const float* __restrict__ W2,
                  float* __restrict__ out) {
  __shared__ float hsh[8][HD][3];
  const int wid = threadIdx.x >> 5, lane = threadIdx.x & 31;
  const int node = blockIdx.x*8 + wid;
  const float* xr = x + (size_t)node*CIN*3;
  float h0 = 0.0f, h1 = 0.0f, h2 = 0.0f;
  for (int c = 0; c < CIN; ++c) {
    const float w = W1[c*HD + lane];
    h0 += xr[c*3+0]*w; h1 += xr[c*3+1]*w; h2 += xr[c*3+2]*w;
  }
  const float nrm = fsqrt(h0*h0 + h1*h1 + h2*h2 + EPSF);
  const float s = swishf(nrm + b1[lane]) * frcp(nrm);
  hsh[wid][lane][0] = h0*s; hsh[wid][lane][1] = h1*s; hsh[wid][lane][2] = h2*s;
  __syncthreads();
  if (lane < DOUT) {
    float o0 = 0.0f, o1 = 0.0f, o2 = 0.0f;
    for (int d = 0; d < HD; ++d) {
      const float w = W2[d*DOUT + lane];
      o0 += hsh[wid][d][0]*w; o1 += hsh[wid][d][1]*w; o2 += hsh[wid][d][2]*w;
    }
    float* op = out + ((size_t)node*DOUT + lane)*3;
    op[0] = o0; op[1] = o1; op[2] = o2;
  }
}

// ---------------------------------------------------------------------------
// out = [vel, scales*vel+trans] ++ scales ++ translations
// ---------------------------------------------------------------------------
__global__ __launch_bounds__(256)
void final_kernel(const float* __restrict__ y, const float* __restrict__ f1b,
                  float* __restrict__ out) {
  const int idx = blockIdx.x*256 + threadIdx.x;
  if (idx >= N_PTS*3) return;
  const int nrow = idx/3, m = idx%3;
  const float vel = y[N_PTS*3 + idx];
  const float sc = f1b[(nrow*2 + 0)*3 + m];
  const float tr = f1b[(nrow*2 + 1)*3 + m];
  out[idx] = vel;
  out[N_PTS*3 + idx] = sc*vel + tr;
  out[2*N_PTS*3 + idx] = sc;
  out[3*N_PTS*3 + idx] = tr;
}

// ---------------------------------------------------------------------------
// Input flattening (JAX pytree = sorted dict keys):
//  0:t 1:y 2:feat0
//  layer0: p00 W1/W2/b1/b2 = 3..6, p01 = 7..10, mlp0 W1/W2/b1/b2 = 11..14,
//          mlp1 W1/W2/b1 = 15..17
//  layer1: p00=18..21 p01=22..25 p10=26..29 p11c=30..33 p11i=34..37,
//          mlp0=38..41, mlp1 W1/W2/b1 = 42..44
//  time:   W1/W2/b1/b2 = 45..48
// ---------------------------------------------------------------------------
extern "C" void kernel_launch(void* const* d_in, const int* in_sizes, int n_in,
                              void* d_out, int out_size, void* d_ws, size_t ws_size,
                              hipStream_t stream) {
  const float* t     = (const float*)d_in[0];
  const float* y     = (const float*)d_in[1];
  const float* feat0 = (const float*)d_in[2];
  auto in = [&](int k) { return (const float*)d_in[k]; };

  // workspace carve-out (~27.5 MB total, all 1KB-aligned offsets)
  char* w = (char*)d_ws;
  _Float16* rbf_h = (_Float16*)(w);          // 768*768*16 f16 = 18,874,368 B
  float* unit = (float*)(w + 18874368);      // 768*768*3  f32 =  7,077,888 B
  float* f0a  = (float*)(w + 25952256);      // 768*32     f32 =     98,304 B
  float* f0b  = (float*)(w + 26050560);      //                      98,304 B
  float* f1a  = (float*)(w + 26148864);      // 768*32*3   f32 =    294,912 B
  float* f1b  = (float*)(w + 26443776);      // 768*2*3    f32 =     18,432 B
  float* c0   = (float*)(w + 26462208);      // 768*64     f32 =    196,608 B
  float* c1   = (float*)(w + 26658816);      // 768*96*3   f32 =    884,736 B

  geom_kernel<<<(N_PTS*N_PTS + 255)/256, 256, 0, stream>>>(y, rbf_h, unit);
  f0_kernel<<<96, 256, 0, stream>>>(t, feat0, in(45), in(46), in(47), in(48), f0a);

  ConvP P0{};
  for (int f = 0; f < 2; ++f) {
    P0.W1[f] = in(3 + 4*f); P0.W2[f] = in(4 + 4*f);
    P0.b1[f] = in(5 + 4*f); P0.b2[f] = in(6 + 4*f);
  }
  conv_kernel<0><<<N_PTS, 256, 0, stream>>>(rbf_h, unit, f0a, (const float*)nullptr,
                                            P0, c0, c1);
  mlpL0_kernel<32><<<96, 256, 0, stream>>>(c0, in(11), in(13), in(12), in(14), f0b);
  mlpL1_kernel<32, 32><<<96, 256, 0, stream>>>(c1, in(15), in(17), in(16), f1a);

  ConvP P1{};
  for (int f = 0; f < 5; ++f) {
    P1.W1[f] = in(18 + 4*f); P1.W2[f] = in(19 + 4*f);
    P1.b1[f] = in(20 + 4*f); P1.b2[f] = in(21 + 4*f);
  }
  conv_kernel<1><<<N_PTS, 256, 0, stream>>>(rbf_h, unit, f0b, f1a, P1, c0, c1);
  // layer-1 mlp0 output (f0) is dead: final output depends only on f1 -> skip.
  mlpL1_kernel<96, 2><<<96, 256, 0, stream>>>(c1, in(42), in(44), in(43), f1b);

  final_kernel<<<(N_PTS*3 + 255)/256, 256, 0, stream>>>(y, f1b, (float*)d_out);

  (void)in_sizes; (void)n_in; (void)out_size; (void)ws_size;
}
